// FCLSTM_55405078118960
// MI455X (gfx1250) — matrix operand order repaired and checked by reading, not compiled, not run on hardware
//
#include <hip/hip_runtime.h>
#include <hip/hip_bf16.h>

typedef __bf16 bf16;
typedef __attribute__((ext_vector_type(16))) __bf16 bf16x16;
typedef __attribute__((ext_vector_type(8)))  __bf16 bf16x8;
typedef __attribute__((ext_vector_type(8)))  float  f32x8;

// LDS (addrspace(3)) typed pointers, built from 32-bit LDS byte addresses.
typedef bf16x8 __attribute__((address_space(3)))*       las_b8_ptr;
typedef const bf16x8 __attribute__((address_space(3)))* las_b8_cptr;

static constexpr int B_  = 512;
static constexpr int T_  = 128;
static constexpr int D_  = 300;
static constexpr int DP_ = 320;   // D padded to multiple of 32 (WMMA K)
static constexpr int H_  = 1024;
static constexpr int G4_ = 4096;  // 4*H
static constexpr int BSTR = 40;   // LDS B-panel row stride (bf16), 80B -> conflict-free frag reads

#if defined(__gfx1250__) && __has_builtin(__builtin_amdgcn_global_load_async_to_lds_b128)
#define USE_ASYNC_LDS 1
typedef int v4i __attribute__((ext_vector_type(4)));
typedef v4i __attribute__((address_space(1)))* gas_v4i_ptr;
typedef v4i __attribute__((address_space(3)))* las_v4i_ptr;

__device__ __forceinline__ gas_v4i_ptr to_gas_v4i(const void* p) {
  return (gas_v4i_ptr)(unsigned long long)p;           // inttoptr into AS1 (non-const)
}
#endif

__device__ __forceinline__ void wait_async_stage() {
#ifdef USE_ASYNC_LDS
#if __has_builtin(__builtin_amdgcn_s_wait_asynccnt)
  __builtin_amdgcn_s_wait_asynccnt(0);
#else
  asm volatile("s_wait_asynccnt 0x0" ::: "memory");
#endif
#endif
}

// ---------------------------------------------------------------------------
// Fragment layouts per CDNA5 ISA 7.12.2 (wave32):
// A (16x32 bf16): lane L<16 -> row M=L, elems K[kb..kb+7] ++ K[kb+16..kb+23];
//                 lane L>=16 -> row M=L-16, chunks shifted by +8.
// B (32x16 bf16): lane L<16 -> col N=L, elems K[kb..kb+7] ++ K[kb+16..kb+23]; lanes>=16: chunks shifted by +8.
// Weights stored row-major W[n][k] so B column n == contiguous row of W.
// ---------------------------------------------------------------------------
union FragU { bf16x16 v; bf16x8 h[2]; };

__device__ __forceinline__ bf16x16 load_a_frag(const bf16* __restrict__ A, int lda,
                                               int mBase, int kb, int lane) {
  const int m  = mBase + (lane & 15);
  const int hi = (lane >> 4) << 3;          // 0 or 8
  const bf16* p = A + (size_t)m * lda + kb + hi;
  FragU u;
  u.h[0] = *(const bf16x8*)(p);
  u.h[1] = *(const bf16x8*)(p + 16);
  return u.v;
}

// B fragment from LDS: base is a 32-bit LDS byte address -> ds_load_b128 x2
__device__ __forceinline__ bf16x16 lds_b_frag(unsigned base, int j, int lane) {
  const unsigned addr =
      base + 2u * (unsigned)((j * 16 + (lane & 15)) * BSTR + ((lane >> 4) << 3));
  FragU u;
  u.h[0] = *(las_b8_cptr)addr;
  u.h[1] = *(las_b8_cptr)(addr + 32);
  return u.v;
}

// Stage a 128-row x 32-K bf16 B panel (8KB) into LDS. 256 threads: each moves 32B.
__device__ __forceinline__ void stage_B(unsigned sbase, const bf16* __restrict__ W, int ldw,
                                        int n0, int kb, int tid) {
  const int row  = tid >> 1;          // 0..127
  const int half = (tid & 1) << 4;    // 0 or 16 (bf16 elems)
  const bf16* g = W + (size_t)(n0 + row) * ldw + kb + half;
  const unsigned laddr = sbase + 2u * (unsigned)(row * BSTR + half);
#ifdef USE_ASYNC_LDS
  __builtin_amdgcn_global_load_async_to_lds_b128(to_gas_v4i(g),     (las_v4i_ptr)laddr,        0, 0);
  __builtin_amdgcn_global_load_async_to_lds_b128(to_gas_v4i(g + 8), (las_v4i_ptr)(laddr + 16), 0, 0);
#else
  *(las_b8_ptr)laddr        = *(const bf16x8*)g;
  *(las_b8_ptr)(laddr + 16) = *(const bf16x8*)(g + 8);
#endif
}

__device__ __forceinline__ f32x8 wmma_bf16(bf16x16 a, bf16x16 b, f32x8 c) {
  return __builtin_amdgcn_wmma_f32_16x16x32_bf16(false, a, false, b, (short)0, c, false, false);
}

// One K-pipelined GEMM contribution: acc += A[mBase..+16][0..K) @ W[n0..n0+128][0..K)^T
__device__ __forceinline__ void gemm_part(const bf16* __restrict__ A, int lda,
                                          const bf16* __restrict__ W, int ldw, int K,
                                          int mBase, int n0, int lane, int tid,
                                          unsigned sb0, unsigned sb1, f32x8 acc[8]) {
  __syncthreads();                       // previous users of the LDS buffers are done
  stage_B(sb0, W, ldw, n0, 0, tid);
  const unsigned bufs[2] = {sb0, sb1};
  int cur = 0;
  for (int kb = 0; kb < K; kb += 32) {
    wait_async_stage();                  // my async panel writes have landed in LDS
    __syncthreads();                     // everyone's panel in; prev compute done
    if (kb + 32 < K) stage_B(bufs[cur ^ 1], W, ldw, n0, kb + 32, tid);
    if (kb + 32 < K)
      __builtin_prefetch(A + (size_t)(mBase + (lane & 15)) * lda + kb + 32, 0, 3);
    const bf16x16 a = load_a_frag(A, lda, mBase, kb, lane);
    const unsigned base = bufs[cur];
    bf16x16 bf[4];
#pragma unroll
    for (int j = 0; j < 4; ++j) bf[j] = lds_b_frag(base, j, lane);
#pragma unroll
    for (int j = 0; j < 4; ++j) acc[j] = wmma_bf16(a, bf[j], acc[j]);
#pragma unroll
    for (int j = 0; j < 4; ++j) bf[j] = lds_b_frag(base, 4 + j, lane);
#pragma unroll
    for (int j = 0; j < 4; ++j) acc[4 + j] = wmma_bf16(a, bf[j], acc[4 + j]);
    cur ^= 1;
  }
}

// gates[512][4096] = bias + A1 @ W1^T (+ A2 @ W2^T)
// Block = 8 waves = 128M x 128N tile; wave = 16M x 128N (8 WMMA accumulators).
// Grid: (512/128)*(4096/128) = 128 blocks.
__global__ __launch_bounds__(256) void lstm_gemm(
    const bf16* __restrict__ A1, int lda1, const bf16* __restrict__ W1, int ldw1, int K1,
    const bf16* __restrict__ A2, int lda2, const bf16* __restrict__ W2, int ldw2, int K2,
    const float* __restrict__ bias, float* __restrict__ out) {
  __shared__ bf16 sB[2][128 * BSTR];     // 2 x 10KB double-buffered B panel
  const unsigned sb0 = (unsigned)(unsigned long long)(void*)&sB[0][0];
  const unsigned sb1 = (unsigned)(unsigned long long)(void*)&sB[1][0];
  const int tid  = threadIdx.x;
  const int lane = tid & 31;
  const int wave = tid >> 5;
  const int bm = blockIdx.x >> 5;        // 4 M blocks
  const int bn = blockIdx.x & 31;        // 32 N blocks
  const int mBase = bm * 128 + wave * 16;
  const int n0    = bn * 128;
  const int nn = lane & 15;

  f32x8 acc[8];
#pragma unroll
  for (int j = 0; j < 8; ++j) {
    const float bv = bias[n0 + j * 16 + nn];
#pragma unroll
    for (int r = 0; r < 8; ++r) acc[j][r] = bv;
  }

  gemm_part(A1, lda1, W1, ldw1, K1, mBase, n0, lane, tid, sb0, sb1, acc);
  if (A2 != nullptr)
    gemm_part(A2, lda2, W2, ldw2, K2, mBase, n0, lane, tid, sb0, sb1, acc);

  // C/D layout: lane<16 -> N=lane, M=r ; lane>=16 -> N=lane-16, M=8+r
  const int mOff = (lane >> 4) << 3;
#pragma unroll
  for (int j = 0; j < 8; ++j) {
    float* o = out + (size_t)(mBase + mOff) * G4_ + n0 + j * 16 + nn;
#pragma unroll
    for (int r = 0; r < 8; ++r) o[(size_t)r * G4_] = acc[j][r];
  }
}

__device__ __forceinline__ float sigmoidf_(float x) { return 1.0f / (1.0f + __expf(-x)); }

// One thread per (m, j) element of the H-wide state.
__global__ __launch_bounds__(256) void lstm_cell(
    const float* __restrict__ gates, float* __restrict__ c,
    bf16* __restrict__ hdst, int ldh, float* __restrict__ accum) {
  const int idx = blockIdx.x * blockDim.x + threadIdx.x;   // 512*1024
  const int m = idx >> 10;
  const int j = idx & 1023;
  const float* g = gates + (size_t)m * G4_;
  const float iv = sigmoidf_(g[j]);
  const float fv = sigmoidf_(g[H_ + j]);
  const float gv = tanhf(g[2 * H_ + j]);
  const float ov = sigmoidf_(g[3 * H_ + j]);
  const float cn = fv * c[idx] + iv * gv;
  const float hn = ov * tanhf(cn);
  c[idx] = cn;
  hdst[(size_t)m * ldh + j] = (bf16)hn;
  if (accum != nullptr) accum[idx] += hn;
}

// ------------------------- prep kernels -------------------------------------
__global__ void k_conv_pad(const float* __restrict__ src, bf16* __restrict__ dst,
                           int Kin, int Kp, long total) {
  const long i = (long)blockIdx.x * blockDim.x + threadIdx.x;
  if (i >= total) return;
  const long n = i / Kp;
  const int  k = (int)(i % Kp);
  dst[i] = (k < Kin) ? (bf16)src[n * Kin + k] : (bf16)0.0f;
}

__global__ void k_conv_w1(const float* __restrict__ wih1, const float* __restrict__ whh1,
                          bf16* __restrict__ dst) {
  const int i = blockIdx.x * blockDim.x + threadIdx.x;   // 4096*2048
  if (i >= G4_ * 2 * H_) return;
  const int n = i >> 11;
  const int k = i & 2047;
  const float v = (k < H_) ? wih1[(size_t)n * H_ + k] : whh1[(size_t)n * H_ + (k - H_)];
  dst[i] = (bf16)v;
}

__global__ void k_conv_x(const float* __restrict__ x, bf16* __restrict__ xb) {
  const long i = (long)blockIdx.x * blockDim.x + threadIdx.x;   // T*B*DP
  if (i >= (long)T_ * B_ * DP_) return;
  const int  jp = (int)(i % DP_);
  const long r  = i / DP_;
  const int  b  = (int)(r % B_);
  const int  t  = (int)(r / B_);
  xb[i] = (jp < D_) ? (bf16)x[((long)b * T_ + t) * D_ + jp] : (bf16)0.0f;
}

__global__ void k_bias(const float* bih0, const float* bhh0,
                       const float* bih1, const float* bhh1,
                       float* b0, float* b1) {
  const int i = blockIdx.x * blockDim.x + threadIdx.x;
  if (i < G4_) { b0[i] = bih0[i] + bhh0[i]; b1[i] = bih1[i] + bhh1[i]; }
}

__global__ void k_zero(float* p, long n) {
  const long i = (long)blockIdx.x * blockDim.x + threadIdx.x;
  if (i < n) p[i] = 0.0f;
}

__global__ void k_decode(const float* __restrict__ accum, const float* __restrict__ Wdec,
                         const float* __restrict__ bdec, float* __restrict__ out) {
  const int m = blockIdx.x * blockDim.x + threadIdx.x;
  if (m >= B_) return;
  float s = 0.0f;
  const float* a = accum + (size_t)m * H_;
  for (int j = 0; j < H_; ++j) s += a[j] * Wdec[j];
  out[m] = s * (1.0f / (float)T_) + bdec[0];
}

// ---------------------------------------------------------------------------
extern "C" void kernel_launch(void* const* d_in, const int* in_sizes, int n_in,
                              void* d_out, int out_size, void* d_ws, size_t ws_size,
                              hipStream_t stream) {
  (void)in_sizes; (void)n_in; (void)out_size; (void)ws_size;
  const float* x     = (const float*)d_in[0];
  const float* Wih0  = (const float*)d_in[1];
  const float* Whh0  = (const float*)d_in[2];
  const float* bih0  = (const float*)d_in[3];
  const float* bhh0  = (const float*)d_in[4];
  const float* Wih1  = (const float*)d_in[5];
  const float* Whh1  = (const float*)d_in[6];
  const float* bih1  = (const float*)d_in[7];
  const float* bhh1  = (const float*)d_in[8];
  const float* Wdec  = (const float*)d_in[9];
  const float* bdec  = (const float*)d_in[10];
  float* out = (float*)d_out;

  char* ws = (char*)d_ws;
  size_t off = 0;
  auto take = [&](size_t bytes) { char* p = ws + off; off += (bytes + 255) & ~(size_t)255; return p; };
  bf16*  Wih0b = (bf16*)take((size_t)G4_ * DP_ * 2);         // [4096][320]
  bf16*  Whh0b = (bf16*)take((size_t)G4_ * H_ * 2);          // [4096][1024]
  bf16*  W1b   = (bf16*)take((size_t)G4_ * 2 * H_ * 2);      // [4096][2048]
  float* b0    = (float*)take((size_t)G4_ * 4);
  float* b1    = (float*)take((size_t)G4_ * 4);
  bf16*  Xb    = (bf16*)take((size_t)T_ * B_ * DP_ * 2);     // [128][512][320]
  bf16*  hcat  = (bf16*)take((size_t)B_ * 2 * H_ * 2);       // [512][2048] (zeroed)
  float* c0    = (float*)take((size_t)B_ * H_ * 4);
  float* c1    = (float*)take((size_t)B_ * H_ * 4);
  float* acc   = (float*)take((size_t)B_ * H_ * 4);
  float* gates = (float*)take((size_t)B_ * G4_ * 4);         // [512][4096]

  const int TPB = 256;
  { long n = (long)G4_ * DP_;
    k_conv_pad<<<(n + TPB - 1) / TPB, TPB, 0, stream>>>(Wih0, Wih0b, D_, DP_, n); }
  { long n = (long)G4_ * H_;
    k_conv_pad<<<(n + TPB - 1) / TPB, TPB, 0, stream>>>(Whh0, Whh0b, H_, H_, n); }
  { long n = (long)G4_ * 2 * H_;
    k_conv_w1<<<(n + TPB - 1) / TPB, TPB, 0, stream>>>(Wih1, Whh1, W1b); }
  k_bias<<<(G4_ + TPB - 1) / TPB, TPB, 0, stream>>>(bih0, bhh0, bih1, bhh1, b0, b1);
  { long n = (long)T_ * B_ * DP_;
    k_conv_x<<<(n + TPB - 1) / TPB, TPB, 0, stream>>>(x, Xb); }
  { long n = ((long)B_ * 2 * H_ * 2 + (long)B_ * H_ * 4 * 3) / 4;  // hcat+c0+c1+acc
    k_zero<<<(n + TPB - 1) / TPB, TPB, 0, stream>>>((float*)hcat, n); }

  const int gemmBlocks = (B_ / 128) * (G4_ / 128);  // 128 blocks of 8 waves
  const int cellBlocks = (B_ * H_) / TPB;

  for (int t = 0; t < T_; ++t) {
    const bf16* xt = Xb + (size_t)t * B_ * DP_;
    // layer0: gates = b0 + x_t @ Wih0^T (K=320) + h0 @ Whh0^T (K=1024)
    lstm_gemm<<<gemmBlocks, TPB, 0, stream>>>(
        xt, DP_, Wih0b, DP_, DP_,
        (const bf16*)hcat, 2 * H_, Whh0b, H_, H_,
        b0, gates);
    lstm_cell<<<cellBlocks, TPB, 0, stream>>>(gates, c0, hcat, 2 * H_, nullptr);
    // layer1: gates = b1 + [h0_new, h1_prev] @ [Wih1;Whh1]^T (K=2048)
    lstm_gemm<<<gemmBlocks, TPB, 0, stream>>>(
        (const bf16*)hcat, 2 * H_, W1b, 2 * H_, 2 * H_,
        nullptr, 0, nullptr, 0, 0,
        b1, gates);
    lstm_cell<<<cellBlocks, TPB, 0, stream>>>(gates, c1, hcat + H_, 2 * H_, acc);
  }

  k_decode<<<(B_ + TPB - 1) / TPB, TPB, 0, stream>>>(acc, Wdec, bdec, out);
}
